// LSTMCell_21088289423624
// MI455X (gfx1250) — compile-verified, matrix operand order checked by reading, not run on hardware
//
#include <hip/hip_runtime.h>
#include <stdint.h>

// ---------------------------------------------------------------------------
// LSTM cell, B=4096, D_IN=D_H=2048.
//   1) pack_xh : fp32 x,h -> bf16 [x|h] (K=4096) in WMMA A-fragment order
//                [mTile16][kStep32][lane][16 bf16]  (contiguous 1KB frags).
//   2) pack_w  : fp32 W_* -> bf16 fused weights in WMMA B-fragment order
//                [nTile16][kStep32][gate][lane][16 bf16]
//                -> all 8 B-frags of one K-step reachable from ONE base
//                   pointer with 24-bit instruction offsets (gates +1..3KB,
//                   second n-tile +512KB), killing per-iteration 64-bit adds.
//   3) lstm_gemm: LDS-free WMMA GEMM; each wave owns a 32x32 patch of all 4
//                gates (16 v8f accums), 128 K-steps x 16 wmma_f32_16x16x32_bf16,
//                fused sigmoid/tanh epilogue -> h_new, c_new.
// bf16 working set (32MB + 64MB) is L2-resident (192MB).
// ---------------------------------------------------------------------------

typedef __bf16 bf16;
typedef __attribute__((ext_vector_type(16))) __bf16 v16bf;
typedef __attribute__((ext_vector_type(8)))  float  v8f;

#define BATCH   4096
#define DH      2048
#define KTOT    4096
#define KSTEPS  128   // KTOT / 32

static __device__ __forceinline__ unsigned short f2bf(float f) {
  union { float f; unsigned u; } v; v.f = f;
  unsigned u = v.u;
  unsigned r = u + 0x7FFFu + ((u >> 16) & 1u);   // round-to-nearest-even
  return (unsigned short)(r >> 16);
}

static __device__ __forceinline__ float fast_sigmoid(float x) {
  return 1.0f / (1.0f + __expf(-x));
}

static __device__ __forceinline__ float fast_tanh(float x) {
  float ax = fabsf(x);
  float t  = __expf(-2.0f * ax);        // t in (0,1], no overflow
  float r  = (1.0f - t) / (1.0f + t);
  return copysignf(r, x);
}

// ---------------------------------------------------------------------------
// Pack x,h into bf16 A-fragments.
// 16B chunk id c: halfSel=c&1, lane=(c>>1)&31, kStep=(c>>6)&127, mTile=c>>13.
// ISA 16-bit A layout per 16x32 tile:
//   lane L (0..15): bytes[0..15]=K 0..7,  bytes[16..31]=K 16..23 (row M=L)
//   lane L+16     : bytes[0..15]=K 8..15, bytes[16..31]=K 24..31 (row M=L)
// ---------------------------------------------------------------------------
__global__ void pack_xh(const float* __restrict__ x, const float* __restrict__ h,
                        bf16* __restrict__ xh) {
  unsigned c       = blockIdx.x * blockDim.x + threadIdx.x;   // 2,097,152 chunks
  unsigned halfSel = c & 1u;
  unsigned lane    = (c >> 1) & 31u;
  unsigned kStep   = (c >> 6) & 127u;
  unsigned mTile   = c >> 13;
  unsigned m = mTile * 16u + (lane & 15u);
  unsigned k = kStep * 32u + halfSel * 16u + ((lane >> 4) & 1u) * 8u;

  const float* src = (k < 2048u) ? (x + (size_t)m * 2048u + k)
                                 : (h + (size_t)m * 2048u + (k - 2048u));
  float4 f0 = *(const float4*)(src + 0);
  float4 f1 = *(const float4*)(src + 4);

  union { unsigned short s[8]; uint4 q; } o;
  o.s[0] = f2bf(f0.x); o.s[1] = f2bf(f0.y); o.s[2] = f2bf(f0.z); o.s[3] = f2bf(f0.w);
  o.s[4] = f2bf(f1.x); o.s[5] = f2bf(f1.y); o.s[6] = f2bf(f1.z); o.s[7] = f2bf(f1.w);
  ((uint4*)xh)[c] = o.q;
}

// ---------------------------------------------------------------------------
// Pack the 8 weight matrices into bf16 B-fragments, gate-interleaved layout:
//   v16bf index = ((nTile*128 + kStep)*4 + gate)*32 + lane
// 16B chunk id c: halfSel=c&1, lane=(c>>1)&31, gate=(c>>6)&3,
//                 kStep=(c>>8)&127, nTile=c>>15.   (4,194,304 chunks)
// B fragment (32Kx16N) mirrors A with N on lanes:
//   lane n (0..15): K 0..7 / 16..23 ; lane n+16: K 8..15 / 24..31.
// Strided global reads (column of W), fully coalesced 16B writes.
// ---------------------------------------------------------------------------
__global__ void pack_w(const float* __restrict__ Wxi, const float* __restrict__ Whi,
                       const float* __restrict__ Wxf, const float* __restrict__ Whf,
                       const float* __restrict__ Wxc, const float* __restrict__ Whc,
                       const float* __restrict__ Wxo, const float* __restrict__ Who,
                       bf16* __restrict__ wp) {
  unsigned c       = blockIdx.x * blockDim.x + threadIdx.x;
  unsigned halfSel = c & 1u;
  unsigned lane    = (c >> 1) & 31u;
  unsigned g       = (c >> 6) & 3u;
  unsigned kStep   = (c >> 8) & 127u;
  unsigned nTile   = c >> 15;
  unsigned n  = nTile * 16u + (lane & 15u);
  unsigned k0 = kStep * 32u + halfSel * 16u + ((lane >> 4) & 1u) * 8u;

  const float* Wx = (g == 0u) ? Wxi : (g == 1u) ? Wxf : (g == 2u) ? Wxc : Wxo;
  const float* Wh = (g == 0u) ? Whi : (g == 1u) ? Whf : (g == 2u) ? Whc : Who;

  union { unsigned short s[8]; uint4 q; } o;
  if (k0 < 2048u) {
    const float* p = Wx + (size_t)k0 * 2048u + n;
#pragma unroll
    for (int j = 0; j < 8; ++j) o.s[j] = f2bf(p[(size_t)j * 2048u]);
  } else {
    const float* p = Wh + (size_t)(k0 - 2048u) * 2048u + n;
#pragma unroll
    for (int j = 0; j < 8; ++j) o.s[j] = f2bf(p[(size_t)j * 2048u]);
  }
  ((uint4*)wp)[c] = o.q;
}

// ---------------------------------------------------------------------------
// Fused GEMM + LSTM epilogue. 256 threads = 8 waves, 2(M) x 4(N) waves,
// each wave computes 32x32 outputs for ALL four gates. All fragment loads
// come from two base pointers with constant instruction offsets.
// ---------------------------------------------------------------------------
__global__ __launch_bounds__(256, 1) void lstm_gemm(
    const bf16* __restrict__ xh, const bf16* __restrict__ wp,
    const float* __restrict__ c_in,
    const float* __restrict__ b_i, const float* __restrict__ b_f,
    const float* __restrict__ b_c, const float* __restrict__ b_o,
    float* __restrict__ h_out, float* __restrict__ c_out) {
  const unsigned tid  = threadIdx.x;
  const unsigned lane = tid & 31u;
  const unsigned wid  = tid >> 5;
  const unsigned wm   = wid & 1u;   // 2 waves in M
  const unsigned wn   = wid >> 1;   // 4 waves in N

  const unsigned mBase = blockIdx.x * 64u + wm * 32u;   // grid.x = 64
  const unsigned nBase = blockIdx.y * 128u + wn * 32u;  // grid.y = 16
  const unsigned mt0 = mBase >> 4;
  const unsigned nt0 = nBase >> 4;

  // A fragment: index (mTile*128 + ks)*32 + lane ; mTile stride = 4096 v16bf.
  const v16bf* __restrict__ aptr = (const v16bf*)xh + (size_t)mt0 * 4096u + lane;
  // B fragment: index ((nTile*128 + ks)*4 + g)*32 + lane ; nTile stride 16384.
  const v16bf* __restrict__ bptr = (const v16bf*)wp + (size_t)nt0 * 16384u + lane;

  const float* bias[4] = {b_i, b_f, b_c, b_o};
  v8f acc[4][2][2];
#pragma unroll
  for (int g = 0; g < 4; ++g) {
#pragma unroll
    for (int j = 0; j < 2; ++j) {
      float bv = bias[g][nBase + (unsigned)j * 16u + (lane & 15u)];
      v8f v;
#pragma unroll
      for (int r = 0; r < 8; ++r) v[r] = bv;
      acc[g][0][j] = v;
      acc[g][1][j] = v;
    }
  }

#pragma unroll 1
  for (unsigned ks = 0; ks < KSTEPS; ++ks) {
    const v16bf* __restrict__ ap = aptr + (size_t)ks * 32u;
    const v16bf* __restrict__ bp = bptr + (size_t)ks * 128u;
    v16bf a0 = ap[0];         // +0
    v16bf a1 = ap[4096];      // +128KB  (imm offset)
#pragma unroll
    for (int g = 0; g < 4; ++g) {
      v16bf bb0 = bp[(unsigned)g * 32u];            // +g*1KB   (imm offset)
      v16bf bb1 = bp[(unsigned)g * 32u + 16384u];   // +512KB   (imm offset)
      acc[g][0][0] = __builtin_amdgcn_wmma_f32_16x16x32_bf16(
          false, a0, false, bb0, (short)0, acc[g][0][0], false, false);
      acc[g][0][1] = __builtin_amdgcn_wmma_f32_16x16x32_bf16(
          false, a0, false, bb1, (short)0, acc[g][0][1], false, false);
      acc[g][1][0] = __builtin_amdgcn_wmma_f32_16x16x32_bf16(
          false, a1, false, bb0, (short)0, acc[g][1][0], false, false);
      acc[g][1][1] = __builtin_amdgcn_wmma_f32_16x16x32_bf16(
          false, a1, false, bb1, (short)0, acc[g][1][1], false, false);
    }
  }

  // Fused LSTM epilogue.  C/D tile layout: VGPR r, lanes 0-15 -> M=r,
  // lanes 16-31 -> M=r+8; N = lane%16.
#pragma unroll
  for (int i = 0; i < 2; ++i) {
#pragma unroll
    for (int j = 0; j < 2; ++j) {
      unsigned nCol  = nBase + (unsigned)j * 16u + (lane & 15u);
      unsigned mRow0 = mBase + (unsigned)i * 16u + ((lane >> 4) & 1u) * 8u;
#pragma unroll
      for (int r = 0; r < 8; ++r) {
        size_t idx = (size_t)(mRow0 + (unsigned)r) * (size_t)DH + nCol;
        float iv = fast_sigmoid(acc[0][i][j][r]);
        float fv = fast_sigmoid(acc[1][i][j][r]);
        float tc = fast_tanh(acc[2][i][j][r]);
        float ov = fast_sigmoid(acc[3][i][j][r]);
        float cn = fv * c_in[idx] + iv * tc;
        float hn = ov * fast_tanh(cn);
        h_out[idx] = hn;
        c_out[idx] = cn;
      }
    }
  }
}

extern "C" void kernel_launch(void* const* d_in, const int* in_sizes, int n_in,
                              void* d_out, int out_size, void* d_ws, size_t ws_size,
                              hipStream_t stream) {
  (void)in_sizes; (void)n_in; (void)out_size; (void)ws_size;
  const float* x   = (const float*)d_in[0];
  const float* h   = (const float*)d_in[1];
  const float* c   = (const float*)d_in[2];
  const float* Wxi = (const float*)d_in[3];
  const float* Whi = (const float*)d_in[4];
  const float* Wxf = (const float*)d_in[5];
  const float* Whf = (const float*)d_in[6];
  const float* Wxo = (const float*)d_in[7];
  const float* Who = (const float*)d_in[8];
  const float* Wxc = (const float*)d_in[9];
  const float* Whc = (const float*)d_in[10];
  const float* bi  = (const float*)d_in[11];
  const float* bf  = (const float*)d_in[12];
  const float* bo  = (const float*)d_in[13];
  const float* bc  = (const float*)d_in[14];

  bf16* xh = (bf16*)d_ws;                                      // 32 MB
  bf16* wp = (bf16*)((char*)d_ws + (size_t)32 * 1024 * 1024);  // 64 MB

  float* h_out = (float*)d_out;
  float* c_out = h_out + (size_t)BATCH * DH;

  pack_xh<<<8192, 256, 0, stream>>>(x, h, xh);
  pack_w<<<16384, 256, 0, stream>>>(Wxi, Whi, Wxf, Whf, Wxc, Whc, Wxo, Who, wp);

  dim3 grid(64, 16);
  lstm_gemm<<<grid, 256, 0, stream>>>(xh, wp, c, bi, bf, bc, bo, h_out, c_out);
}
